// MambaMixer_57672820851296
// MI455X (gfx1250) — compile-verified, matrix operand order checked by reading, not run on hardware
//
#include <hip/hip_runtime.h>
#include <hip/hip_bf16.h>

typedef __bf16 bf16;
typedef __attribute__((ext_vector_type(16))) __bf16 v16bf;
typedef __attribute__((ext_vector_type(8)))  float  v8f;

// CDNA5 async global->LDS staging (ASYNCcnt-tracked), if toolchain declares it.
#if defined(__has_builtin)
#if __has_builtin(__builtin_amdgcn_global_load_async_to_lds_b128) && \
    __has_builtin(__builtin_amdgcn_s_wait_asynccnt)
#define HAVE_ASYNC_LDS 1
#endif
#endif
#ifndef HAVE_ASYNC_LDS
#define HAVE_ASYNC_LDS 0
#endif

#if HAVE_ASYNC_LDS
typedef __attribute__((__vector_size__(16))) int v4i_t;
__device__ __forceinline__ void async_copy16(const void* g, void* l) {
  __builtin_amdgcn_global_load_async_to_lds_b128(
      (__attribute__((address_space(1))) v4i_t*)g,
      (__attribute__((address_space(3))) v4i_t*)l, 0, 0);
}
__device__ __forceinline__ void async_wait0() {
  __builtin_amdgcn_s_wait_asynccnt(0);
}
#endif

// ---------------------------------------------------------------------------
// WMMA fragment helpers (CDNA5 wave32, 16x16x32 bf16 -> f32)
// A (16x32, row M = lane&15): VGPR v, half h=lane>>4, packed t:
//   K = (v&3)*2 + (v>>2)*16 + 8*h + t
// B (32x16): column N = lane&15, same K mapping over VGPR/half.
// C/D (16x16 f32): M = v + 8*h, N = lane&15.
// ---------------------------------------------------------------------------
__device__ __forceinline__ int frag_k(int i, int h) {
  int v = i >> 1, t = i & 1;
  return ((v & 3) << 1) + ((v >> 2) << 4) + (h << 3) + t;
}

template <typename F>
__device__ __forceinline__ v16bf make_frag(int h, F f) {
  v16bf r;
#pragma unroll
  for (int i = 0; i < 16; ++i) r[i] = f(frag_k(i, h));
  return r;
}

__device__ __forceinline__ v8f wmma_bf16(v16bf a, v16bf b, v8f c) {
  return __builtin_amdgcn_wmma_f32_16x16x32_bf16(
      /*neg_a=*/false, a, /*neg_b=*/false, b,
      /*c_mod=*/(short)0, c, /*reuse_a=*/false, /*reuse_b=*/false);
}

__device__ __forceinline__ float silu_f(float x) {
  return x / (1.f + __expf(-x));
}

// Model dims
#define SEQ    2048
#define HID    2048
#define INTER  4096
#define GN     128
#define NH     64
#define HD     64
#define CONVD  4352
#define D_IN   8512
#define L_CHK  256
#define NCHK   8

// ---------------------------------------------------------------------------
// f32 -> bf16 conversion
// ---------------------------------------------------------------------------
__global__ __launch_bounds__(256) void cvt_bf16_kernel(const float* __restrict__ src,
                                                       bf16* __restrict__ dst, int n) {
  int g = blockIdx.x * 256 + threadIdx.x;
  if (g < n) dst[g] = (bf16)src[g];
}

// ---------------------------------------------------------------------------
// Tiled bf16 WMMA GEMM: C[M,N] = A[M,K] * B[K,N], A,B row-major bf16.
// Block = 256 threads (8 waves, 4x2), tile 128(M) x 64(N), K-step 32.
// Tiles staged to LDS via CDNA5 async-to-LDS when available.
// MODE 0: plain f32 store.  MODE 1: split into gate(bf16)/xBC(f32)/dt(f32).
// ---------------------------------------------------------------------------
template <int MODE>
__global__ __launch_bounds__(256) void gemm_bf16_kernel(
    const bf16* __restrict__ A, const bf16* __restrict__ B,
    int M, int N, int K,
    float* __restrict__ outF, bf16* __restrict__ gateb,
    float* __restrict__ xbc, float* __restrict__ dtraw) {
  __shared__ __align__(16) bf16 sA[128 * 32];
  __shared__ __align__(16) bf16 sB[32 * 64];
  const int tid = threadIdx.x;
  const int wid = tid >> 5, lane = tid & 31;
  const int h = lane >> 4, idx = lane & 15;
  const int wm = wid & 3, wn = wid >> 2;  // 4 x 2 wave grid, 32x32 per wave
  const int mBase = blockIdx.y * 128, nBase = blockIdx.x * 64;

  const int rowA = tid >> 1, cA0 = (tid & 1) * 16;   // A: 2 thr/row, 16 bf16 each
  const int rowB = tid >> 3, cB0 = (tid & 7) * 8;    // B: 8 thr/row, 8 bf16 each

  v8f acc[2][2] = {};
  for (int k0 = 0; k0 < K; k0 += 32) {
    __syncthreads();  // previous iteration done reading LDS
    {
      const bf16* srcA = A + (size_t)(mBase + rowA) * K + k0 + cA0;
      const bf16* srcB = B + (size_t)(k0 + rowB) * N + nBase + cB0;
#if HAVE_ASYNC_LDS
      async_copy16(srcA,     sA + rowA * 32 + cA0);
      async_copy16(srcA + 8, sA + rowA * 32 + cA0 + 8);
      async_copy16(srcB,     sB + rowB * 64 + cB0);
#else
      int4* dA = (int4*)(sA + rowA * 32 + cA0);
      dA[0] = ((const int4*)srcA)[0];
      dA[1] = ((const int4*)srcA)[1];
      *(int4*)(sB + rowB * 64 + cB0) = *(const int4*)srcB;
#endif
      if (k0 + 32 < K) {  // prefetch next K tiles toward L2
        __builtin_prefetch(srcA + 32, 0, 1);
        __builtin_prefetch(srcB + (size_t)32 * N, 0, 1);
      }
    }
#if HAVE_ASYNC_LDS
    async_wait0();
#endif
    __syncthreads();

    v16bf af[2], bfm[2];
#pragma unroll
    for (int sm = 0; sm < 2; ++sm) {
      int r = wm * 32 + sm * 16 + idx;
      af[sm] = make_frag(h, [&](int k) { return sA[r * 32 + k]; });
    }
#pragma unroll
    for (int sn = 0; sn < 2; ++sn) {
      int colB = wn * 32 + sn * 16 + idx;
      bfm[sn] = make_frag(h, [&](int k) { return sB[k * 64 + colB]; });
    }
#pragma unroll
    for (int sm = 0; sm < 2; ++sm)
#pragma unroll
      for (int sn = 0; sn < 2; ++sn)
        acc[sm][sn] = wmma_bf16(af[sm], bfm[sn], acc[sm][sn]);
  }

#pragma unroll
  for (int sm = 0; sm < 2; ++sm)
#pragma unroll
    for (int sn = 0; sn < 2; ++sn)
#pragma unroll
      for (int v = 0; v < 8; ++v) {
        int r = mBase + wm * 32 + sm * 16 + v + 8 * h;
        int c = nBase + wn * 32 + sn * 16 + idx;
        float val = acc[sm][sn][v];
        if constexpr (MODE == 0) {
          outF[(size_t)r * N + c] = val;
        } else {
          if (c < INTER)            gateb[(size_t)r * INTER + c] = (bf16)val;
          else if (c < INTER+CONVD) xbc[(size_t)r * CONVD + (c - INTER)] = val;
          else                      dtraw[r * NH + (c - INTER - CONVD)] = val;
        }
      }
}

// ---------------------------------------------------------------------------
// Causal depthwise conv1d (K=4) + SiLU, split into x/B/C (bf16)
// ---------------------------------------------------------------------------
__global__ __launch_bounds__(256) void conv_silu_kernel(
    const float* __restrict__ xbc, const float* __restrict__ cw,
    const float* __restrict__ cbias, bf16* __restrict__ xb,
    bf16* __restrict__ Bb, bf16* __restrict__ Cb) {
  int g = blockIdx.x * 256 + threadIdx.x;
  if (g >= SEQ * CONVD) return;
  int c = g % CONVD, s = g / CONVD;
  float acc = cbias[c];
#pragma unroll
  for (int k = 0; k < 4; ++k) {
    int sp = s + k - 3;
    if (sp >= 0) acc += cw[c * 4 + k] * xbc[(size_t)sp * CONVD + c];
  }
  float v = silu_f(acc);
  if (c < INTER)          xb[(size_t)s * INTER + c] = (bf16)v;
  else if (c < INTER+GN)  Bb[s * GN + (c - INTER)] = (bf16)v;
  else                    Cb[s * GN + (c - INTER - GN)] = (bf16)v;
}

// ---------------------------------------------------------------------------
// dt softplus + within-chunk cumsum of dt*A + state weights
// one thread per (chunk, head), serial over L=256
// ---------------------------------------------------------------------------
__global__ __launch_bounds__(256) void dtcum_kernel(
    const float* __restrict__ dtraw, const float* __restrict__ dt_bias,
    const float* __restrict__ Av, float* __restrict__ dt_sp,
    float* __restrict__ cA, float* __restrict__ wcoef) {
  int g = blockIdx.x * 256 + threadIdx.x;
  if (g >= NCHK * NH) return;
  int head = g & (NH - 1), chunk = g >> 6;
  float a = Av[head], bias = dt_bias[head], run = 0.f;
  for (int l = 0; l < L_CHK; ++l) {
    int s = chunk * L_CHK + l;
    float x = dtraw[s * NH + head] + bias;
    float sp = (x > 20.f) ? x : log1pf(__expf(x));
    run += sp * a;
    dt_sp[s * NH + head] = sp;
    cA[s * NH + head] = run;
  }
  float tot = run;
  for (int l = 0; l < L_CHK; ++l) {
    int s = chunk * L_CHK + l;
    wcoef[s * NH + head] = __expf(tot - cA[s * NH + head]) * dt_sp[s * NH + head];
  }
}

// ---------------------------------------------------------------------------
// Fused intra-chunk: block per (chunk, head).
//   per s-tile(32): CB = C*B^T via WMMA (K=128); apply causal decay mask
//   at fragment level; round-trip bf16 M-tile through LDS; Y += M * x.
// Epilogue adds D*x skip and writes f32 Y.
// ---------------------------------------------------------------------------
__global__ __launch_bounds__(256) void y_intra_kernel(
    const bf16* __restrict__ Cb, const bf16* __restrict__ Bb,
    const bf16* __restrict__ xb, const float* __restrict__ cA,
    const float* __restrict__ dt_sp, const float* __restrict__ Dv,
    float* __restrict__ Y) {
  const int chunk = blockIdx.x >> 6, head = blockIdx.x & 63;
  __shared__ __align__(16) bf16 sM[L_CHK * 32];
  __shared__ __align__(16) bf16 sX[32 * HD];
  __shared__ float sCA[L_CHK], sDT[L_CHK];
  const int tid = threadIdx.x, wid = tid >> 5, lane = tid & 31;
  const int h = lane >> 4, idx = lane & 15;
  const int s0 = chunk * L_CHK;

  sCA[tid] = cA[(s0 + tid) * NH + head];
  sDT[tid] = dt_sp[(s0 + tid) * NH + head];
  __syncthreads();

  v8f yacc[2][4] = {};
  for (int st = 0; st < 8; ++st) {
    const int sb = st * 32;
    // ---- CB = C * B^T on this wave's 32 rows x 32 cols, K = 128 ----
    v8f cbv[2][2] = {};
#pragma unroll
    for (int kk = 0; kk < 4; ++kk) {
      const int nb = kk * 32;
      v16bf af[2], bfm[2];
#pragma unroll
      for (int ls = 0; ls < 2; ++ls) {
        int rowl = wid * 32 + ls * 16 + idx;
        af[ls] = make_frag(h, [&](int k) {
          return Cb[(size_t)(s0 + rowl) * GN + nb + k];
        });
      }
#pragma unroll
      for (int ss = 0; ss < 2; ++ss) {
        int rows = sb + ss * 16 + idx;
        bfm[ss] = make_frag(h, [&](int k) {
          return Bb[(size_t)(s0 + rows) * GN + nb + k];
        });
      }
#pragma unroll
      for (int ls = 0; ls < 2; ++ls)
#pragma unroll
        for (int ss = 0; ss < 2; ++ss)
          cbv[ls][ss] = wmma_bf16(af[ls], bfm[ss], cbv[ls][ss]);
    }

    __syncthreads();  // previous Y-phase done reading sM/sX
    {  // stage x tile 32 x 64 for this head (async when available)
      int r = tid >> 3, c0 = (tid & 7) * 8;
      const bf16* src = xb + (size_t)(s0 + sb + r) * INTER + head * HD + c0;
#if HAVE_ASYNC_LDS
      async_copy16(src, sX + r * HD + c0);
#else
      *(int4*)(sX + r * HD + c0) = *(const int4*)src;
#endif
    }
    // ---- mask M[l,s] = CB * exp(cA_l - cA_s) * dt_s (s<=l), store bf16 ----
#pragma unroll
    for (int ls = 0; ls < 2; ++ls)
#pragma unroll
      for (int ss = 0; ss < 2; ++ss)
#pragma unroll
        for (int v = 0; v < 8; ++v) {
          int l = wid * 32 + ls * 16 + v + 8 * h;
          int s = sb + ss * 16 + idx;
          float m = 0.f;
          if (s <= l) m = cbv[ls][ss][v] * __expf(sCA[l] - sCA[s]) * sDT[s];
          sM[l * 32 + (s - sb)] = (bf16)m;
        }
#if HAVE_ASYNC_LDS
    async_wait0();
#endif
    __syncthreads();

    // ---- Y += M * x  (K = 32 this s-tile) ----
    v16bf bx[4];
#pragma unroll
    for (int ps = 0; ps < 4; ++ps) {
      int col = ps * 16 + idx;
      bx[ps] = make_frag(h, [&](int k) { return sX[k * HD + col]; });
    }
#pragma unroll
    for (int ls = 0; ls < 2; ++ls) {
      int rowl = wid * 32 + ls * 16 + idx;
      v16bf am = make_frag(h, [&](int k) { return sM[rowl * 32 + k]; });
#pragma unroll
      for (int ps = 0; ps < 4; ++ps)
        yacc[ls][ps] = wmma_bf16(am, bx[ps], yacc[ls][ps]);
    }
  }

  const float Dh = Dv[head];
#pragma unroll
  for (int ls = 0; ls < 2; ++ls)
#pragma unroll
    for (int ps = 0; ps < 4; ++ps)
#pragma unroll
      for (int v = 0; v < 8; ++v) {
        int l = s0 + wid * 32 + ls * 16 + v + 8 * h;
        int c = head * HD + ps * 16 + idx;
        size_t o = (size_t)l * INTER + c;
        Y[o] = yacc[ls][ps][v] + Dh * (float)xb[o];
      }
}

// ---------------------------------------------------------------------------
// Chunk end-state contributions: states[c,h,p,n] = sum_s w[s]*x[s,p]*B[s,n]
// block per (chunk, head); output 64(p) x 128(n); K=256 over s.
// ---------------------------------------------------------------------------
__global__ __launch_bounds__(256) void states_kernel(
    const bf16* __restrict__ xb, const bf16* __restrict__ Bb,
    const float* __restrict__ wcoef, float* __restrict__ states) {
  const int chunk = blockIdx.x >> 6, head = blockIdx.x & 63;
  __shared__ float sW[L_CHK];
  const int tid = threadIdx.x, wid = tid >> 5, lane = tid & 31;
  const int h = lane >> 4, idx = lane & 15;
  const int wm = wid & 1, wn = wid >> 1;  // 2(p) x 4(n) waves
  const int s0 = chunk * L_CHK;
  sW[tid] = wcoef[(s0 + tid) * NH + head];
  __syncthreads();

  v8f acc[2][2] = {};
  for (int k0 = 0; k0 < L_CHK; k0 += 32) {
    v16bf af[2], bfm[2];
#pragma unroll
    for (int sm = 0; sm < 2; ++sm) {
      int p = wm * 32 + sm * 16 + idx;
      af[sm] = make_frag(h, [&](int k) {
        float xv = (float)xb[(size_t)(s0 + k0 + k) * INTER + head * HD + p];
        return (bf16)(xv * sW[k0 + k]);
      });
    }
#pragma unroll
    for (int sn = 0; sn < 2; ++sn) {
      int n = wn * 32 + sn * 16 + idx;
      bfm[sn] = make_frag(h, [&](int k) {
        return Bb[(size_t)(s0 + k0 + k) * GN + n];
      });
    }
#pragma unroll
    for (int sm = 0; sm < 2; ++sm)
#pragma unroll
      for (int sn = 0; sn < 2; ++sn)
        acc[sm][sn] = wmma_bf16(af[sm], bfm[sn], acc[sm][sn]);
  }
#pragma unroll
  for (int sm = 0; sm < 2; ++sm)
#pragma unroll
    for (int sn = 0; sn < 2; ++sn)
#pragma unroll
      for (int v = 0; v < 8; ++v) {
        int p = wm * 32 + sm * 16 + v + 8 * h;
        int n = wn * 32 + sn * 16 + idx;
        states[(((size_t)chunk * NH + head) * HD + p) * GN + n] = acc[sm][sn][v];
      }
}

// ---------------------------------------------------------------------------
// Sequential scan over 8 chunks; emits state ENTERING each chunk (bf16).
// ---------------------------------------------------------------------------
__global__ __launch_bounds__(256) void scan_kernel(
    const float* __restrict__ states, const float* __restrict__ cA,
    bf16* __restrict__ prevb) {
  int gid = blockIdx.x * 256 + threadIdx.x;  // NH*HD*GN = 524288
  int n = gid & (GN - 1), p = (gid >> 7) & (HD - 1), head = gid >> 13;
  float carry = 0.f;
  for (int c = 0; c < NCHK; ++c) {
    size_t o = (((size_t)c * NH + head) * HD + p) * GN + n;
    prevb[o] = (bf16)carry;
    float cd = __expf(cA[(c * L_CHK + L_CHK - 1) * NH + head]);
    carry = carry * cd + states[o];
  }
}

// ---------------------------------------------------------------------------
// Inter-chunk: Y += exp(cA[l]) * (C[l,:] . prev[p,:]); block per (chunk,head)
// ---------------------------------------------------------------------------
__global__ __launch_bounds__(256) void y_inter_kernel(
    const bf16* __restrict__ Cb, const bf16* __restrict__ prevb,
    const float* __restrict__ cA, float* __restrict__ Y) {
  const int chunk = blockIdx.x >> 6, head = blockIdx.x & 63;
  __shared__ float sCA[L_CHK];
  const int tid = threadIdx.x, wid = tid >> 5, lane = tid & 31;
  const int h = lane >> 4, idx = lane & 15;
  const int s0 = chunk * L_CHK;
  sCA[tid] = cA[(s0 + tid) * NH + head];
  __syncthreads();

  const bf16* pv = prevb + ((size_t)chunk * NH + head) * HD * GN;
  v8f acc[2][4] = {};
  for (int k0 = 0; k0 < GN; k0 += 32) {
    v16bf af[2], bfm[4];
#pragma unroll
    for (int ls = 0; ls < 2; ++ls) {
      int rowl = wid * 32 + ls * 16 + idx;
      af[ls] = make_frag(h, [&](int k) {
        return Cb[(size_t)(s0 + rowl) * GN + k0 + k];
      });
    }
#pragma unroll
    for (int ps = 0; ps < 4; ++ps) {
      int p = ps * 16 + idx;
      bfm[ps] = make_frag(h, [&](int k) { return pv[(size_t)p * GN + k0 + k]; });
    }
#pragma unroll
    for (int ls = 0; ls < 2; ++ls)
#pragma unroll
      for (int ps = 0; ps < 4; ++ps)
        acc[ls][ps] = wmma_bf16(af[ls], bfm[ps], acc[ls][ps]);
  }
#pragma unroll
  for (int ls = 0; ls < 2; ++ls)
#pragma unroll
    for (int ps = 0; ps < 4; ++ps)
#pragma unroll
      for (int v = 0; v < 8; ++v) {
        int l = wid * 32 + ls * 16 + v + 8 * h;
        int c = head * HD + ps * 16 + idx;
        size_t o = (size_t)(s0 + l) * INTER + c;
        Y[o] += __expf(sCA[l]) * acc[ls][ps][v];
      }
}

// ---------------------------------------------------------------------------
// Gated RMSNorm (f32), write z as bf16. Block per row of 4096.
// ---------------------------------------------------------------------------
__global__ __launch_bounds__(256) void norm_kernel(
    const float* __restrict__ Y, const bf16* __restrict__ gateb,
    const float* __restrict__ nw, bf16* __restrict__ zb) {
  const int row = blockIdx.x, tid = threadIdx.x;
  float zv[16], ss = 0.f;
#pragma unroll
  for (int i = 0; i < 16; ++i) {
    int c = tid + i * 256;
    float g = (float)gateb[(size_t)row * INTER + c];
    float z = Y[(size_t)row * INTER + c] * silu_f(g);
    zv[i] = z;
    ss += z * z;
  }
  __shared__ float red[256];
  red[tid] = ss;
  __syncthreads();
  for (int off = 128; off > 0; off >>= 1) {
    if (tid < off) red[tid] += red[tid + off];
    __syncthreads();
  }
  float rs = rsqrtf(red[0] / (float)INTER + 1e-5f);
#pragma unroll
  for (int i = 0; i < 16; ++i) {
    int c = tid + i * 256;
    zb[(size_t)row * INTER + c] = (bf16)(zv[i] * rs * nw[c]);
  }
}

// ---------------------------------------------------------------------------
extern "C" void kernel_launch(void* const* d_in, const int* in_sizes, int n_in,
                              void* d_out, int out_size, void* d_ws, size_t ws_size,
                              hipStream_t stream) {
  const float* hidden  = (const float*)d_in[0];
  const float* W_in    = (const float*)d_in[1];
  const float* conv_w  = (const float*)d_in[2];
  const float* conv_b  = (const float*)d_in[3];
  const float* dt_bias = (const float*)d_in[4];
  const float* Av      = (const float*)d_in[5];
  const float* Dv      = (const float*)d_in[6];
  const float* norm_w  = (const float*)d_in[7];
  const float* W_out   = (const float*)d_in[8];
  float* out = (float*)d_out;

  char* ws = (char*)d_ws;
  size_t off = 0;
  auto alloc = [&](size_t bytes) -> void* {
    void* p = ws + off;
    off = (off + bytes + 255) & ~(size_t)255;
    return p;
  };
  bf16*  hb    = (bf16*)alloc((size_t)SEQ * HID * 2);        //  8.4 MB
  bf16*  wib   = (bf16*)alloc((size_t)HID * D_IN * 2);       // 34.9 MB (reused below)
  bf16*  wob   = (bf16*)alloc((size_t)INTER * HID * 2);      // 16.8 MB
  bf16*  gateb = (bf16*)alloc((size_t)SEQ * INTER * 2);      // 16.8 MB
  float* xbc   = (float*)alloc((size_t)SEQ * CONVD * 4);     // 35.7 MB (Y reuses)
  float* dtraw = (float*)alloc((size_t)SEQ * NH * 4);
  bf16*  xb    = (bf16*)alloc((size_t)SEQ * INTER * 2);      // 16.8 MB
  bf16*  Bb    = (bf16*)alloc((size_t)SEQ * GN * 2);
  bf16*  Cb    = (bf16*)alloc((size_t)SEQ * GN * 2);
  float* dtsp  = (float*)alloc((size_t)SEQ * NH * 4);
  float* cA    = (float*)alloc((size_t)SEQ * NH * 4);
  float* wcoef = (float*)alloc((size_t)SEQ * NH * 4);
  bf16*  zb    = (bf16*)alloc((size_t)SEQ * INTER * 2);      // 16.8 MB
  // aliases: states+prev reuse wib (dead after gemm_in); Y reuses xbc (dead after conv)
  float* states = (float*)wib;                               // 16.8 MB
  bf16*  prevb  = (bf16*)((char*)wib + (size_t)NCHK * NH * HD * GN * 4);  // 8.4 MB
  float* Y      = xbc;                                       // 33.6 MB

  // 1) downconvert weights/activations to bf16
  cvt_bf16_kernel<<<(SEQ * HID + 255) / 256, 256, 0, stream>>>(hidden, hb, SEQ * HID);
  cvt_bf16_kernel<<<(HID * D_IN + 255) / 256, 256, 0, stream>>>(W_in, wib, HID * D_IN);
  cvt_bf16_kernel<<<(INTER * HID + 255) / 256, 256, 0, stream>>>(W_out, wob, INTER * HID);

  // 2) in_proj GEMM (2048x8512x2048), split epilogue
  gemm_bf16_kernel<1><<<dim3(D_IN / 64, SEQ / 128), 256, 0, stream>>>(
      hb, wib, SEQ, D_IN, HID, nullptr, gateb, xbc, dtraw);

  // 3) conv + SiLU, split x/B/C
  conv_silu_kernel<<<(SEQ * CONVD + 255) / 256, 256, 0, stream>>>(
      xbc, conv_w, conv_b, xb, Bb, Cb);

  // 4) dt softplus, cumsum(dt*A), state weights
  dtcum_kernel<<<(NCHK * NH + 255) / 256, 256, 0, stream>>>(
      dtraw, dt_bias, Av, dtsp, cA, wcoef);

  // 5) fused intra-chunk attention-like pass (writes Y, overwrites xbc)
  y_intra_kernel<<<NCHK * NH, 256, 0, stream>>>(Cb, Bb, xb, cA, dtsp, Dv, Y);

  // 6) per-chunk end states
  states_kernel<<<NCHK * NH, 256, 0, stream>>>(xb, Bb, wcoef, states);

  // 7) chunk scan -> states entering each chunk
  scan_kernel<<<(NH * HD * GN + 255) / 256, 256, 0, stream>>>(states, cA, prevb);

  // 8) inter-chunk contribution, Y +=
  y_inter_kernel<<<NCHK * NH, 256, 0, stream>>>(Cb, prevb, cA, Y);

  // 9) gated RMSNorm -> z bf16
  norm_kernel<<<SEQ, 256, 0, stream>>>(Y, gateb, norm_w, zb);

  // 10) out_proj GEMM (2048x2048x4096) -> f32 output
  gemm_bf16_kernel<0><<<dim3(HID / 64, SEQ / 128), 256, 0, stream>>>(
      zb, wob, SEQ, HID, INTER, out, nullptr, nullptr, nullptr);
}
